// RGAST2_30562987278619
// MI455X (gfx1250) — matrix operand order repaired and checked by reading, not compiled, not run on hardware
//
#include <hip/hip_runtime.h>

// ---------- types for WMMA ----------
typedef __attribute__((ext_vector_type(16))) __bf16 v16bf;
typedef __attribute__((ext_vector_type(8)))  float  v8f;
typedef unsigned short ushort_t;

union V16BF {
    v16bf v;
    uint4 q[2];   // 2 x 16B = 32B = 16 bf16
};

// round-to-nearest-even float -> bf16 bit pattern
__device__ __forceinline__ unsigned short f2bf(float f) {
    unsigned int x = __float_as_uint(f);
    x = (x + 0x7FFFu + ((x >> 16) & 1u)) >> 16;
    return (unsigned short)x;
}

// float atomic max via signed/unsigned int ordering trick
__device__ __forceinline__ void atomicMaxFloat(float* addr, float val) {
    if (val >= 0.0f)
        atomicMax((int*)addr, __float_as_int(val));
    else
        atomicMin((unsigned int*)addr, __float_as_uint(val));
}

// =====================================================================
// streaming fp32 -> bf16 (packed x2)
// =====================================================================
__global__ void rgat_f32_to_bf16x2(const float2* __restrict__ src,
                                   unsigned int* __restrict__ dst, long n2) {
    long i = (long)blockIdx.x * blockDim.x + threadIdx.x;
    if (i >= n2) return;
    float2 v = src[i];
    dst[i] = (unsigned int)f2bf(v.x) | ((unsigned int)f2bf(v.y) << 16);
}

// transpose + convert weights: Wt[z][o][i] = bf16( W[z][i][o] )
__global__ void rgat_wt_bf16(const float* __restrict__ W,
                             ushort_t* __restrict__ Wt,
                             int in_dim, int out_dim, long total) {
    long i = (long)blockIdx.x * blockDim.x + threadIdx.x;
    if (i >= total) return;
    long per = (long)in_dim * out_dim;
    long z = i / per;
    long rem = i - z * per;
    int o = (int)(rem / in_dim);
    int ii = (int)(rem - (long)o * in_dim);
    Wt[i] = f2bf(W[z * per + (size_t)ii * out_dim + o]);
}

// =====================================================================
// WMMA GEMM:  Y[z] = act( X @ W[z] + bias )
// X pre-converted bf16 [rows, IN_DIM]; Wt bf16 transposed [z][out][IN_DIM].
// One wave per 16x16 tile; K fully unrolled (IN_DIM compile-time).
// act: 0 = none, 1 = relu
// =====================================================================
template <int IN_DIM>
__global__ void rgat_wmma_gemm(const ushort_t* __restrict__ Xbf,
                               const ushort_t* __restrict__ Wtbase,
                               const float* __restrict__ bias,
                               float* __restrict__ Ybase,
                               int out_dim, long wt_stride, long y_stride, int act) {
    const ushort_t* Wt = Wtbase + (long)blockIdx.z * wt_stride;
    float*          Y  = Ybase  + (long)blockIdx.z * y_stride;

    const int lane  = threadIdx.x;          // 0..31
    const int mbase = blockIdx.x * 16;      // node tile
    const int nbase = blockIdx.y * 16;      // output tile
    const int l15   = lane & 15;            // row of A / col of B
    const bool lo   = lane < 16;

    const ushort_t* xrow = Xbf + (size_t)(mbase + l15) * IN_DIM + (lo ? 0 : 8);
    const ushort_t* wrow = Wt  + (size_t)(nbase + l15) * IN_DIM + (lo ? 0 : 8);

    v8f c = {};
#pragma unroll
    for (int k0 = 0; k0 < IN_DIM; k0 += 32) {
        // A: lane<16 -> K {k0+0..7, k0+16..23}; lane>=16 -> {k0+8..15, k0+24..31}
        V16BF A;
        A.q[0] = *(const uint4*)(xrow + k0);
        A.q[1] = *(const uint4*)(xrow + k0 + 16);
        // B: lane<16 -> K {k0+0..7, k0+16..23} of column nbase+l15; lane>=16 -> {k0+8..15, k0+24..31}
        V16BF B;
        B.q[0] = *(const uint4*)(wrow + k0);
        B.q[1] = *(const uint4*)(wrow + k0 + 16);

        c = __builtin_amdgcn_wmma_f32_16x16x32_bf16(
                false, A.v, false, B.v, (short)0, c, false, false);
    }

    // epilogue: C/D layout, lane<16 -> M=r, lane>=16 -> M=8+r
    const int mofs = lo ? 0 : 8;
    const float bb = bias ? bias[nbase + l15] : 0.0f;
#pragma unroll
    for (int r = 0; r < 8; ++r) {
        float v = c[r] + bb;
        if (act == 1) v = fmaxf(v, 0.0f);
        Y[(size_t)(mbase + mofs + r) * out_dim + nbase + l15] = v;
    }
}

// =====================================================================
// per-(relation,node) query/key scores: qn[i] = xw_row . q ; kn likewise
// =====================================================================
__global__ void rgat_qk(const float* __restrict__ xw,
                        const float* __restrict__ q,
                        const float* __restrict__ k,
                        float* __restrict__ qn, float* __restrict__ kn,
                        int total, int out_dim) {
    int i = blockIdx.x * blockDim.x + threadIdx.x;
    if (i >= total) return;
    const float* row = xw + (size_t)i * out_dim;
    float aq = 0.0f, ak = 0.0f;
    for (int o = 0; o < out_dim; ++o) {
        float v = row[o];
        aq += v * q[o];
        ak += v * k[o];
    }
    qn[i] = aq;
    kn[i] = ak;
}

__global__ void rgat_fill(float* __restrict__ p, float v, long n) {
    long i = (long)blockIdx.x * blockDim.x + threadIdx.x;
    if (i < n) p[i] = v;
}

// edge logits + segment max
__global__ void rgat_edge_logits(const int* __restrict__ src, const int* __restrict__ dst,
                                 const int* __restrict__ et,
                                 const float* __restrict__ qn, const float* __restrict__ kn,
                                 float* __restrict__ ea, float* __restrict__ m,
                                 int E, int N) {
    int e = blockIdx.x * blockDim.x + threadIdx.x;
    if (e >= E) return;
    int d = dst[e], s = src[e], t = et[e];
    float a = qn[t * N + d] + kn[t * N + s];
    a = (a >= 0.0f) ? a : 0.2f * a;          // leaky_relu(0.2)
    ea[e] = a;
    atomicMaxFloat(&m[d], a);
}

// exp(a - m[dst]) + segment sum (in-place on edge buffer)
__global__ void rgat_edge_exp(const int* __restrict__ dst,
                              float* __restrict__ ea,
                              const float* __restrict__ m,
                              float* __restrict__ ssum, int E) {
    int e = blockIdx.x * blockDim.x + threadIdx.x;
    if (e >= E) return;
    int d = dst[e];
    float ex = __expf(ea[e] - m[d]);
    ea[e] = ex;
    atomicAdd(&ssum[d], ex);
}

// alpha * xw[etype,src] scatter-added into agg[dst]; LPE lanes per edge,
// each lane moves a float2.  out_dim == 2*LPE.
template <int LPE>
__global__ void rgat_edge_scatter(const int* __restrict__ src, const int* __restrict__ dst,
                                  const int* __restrict__ et,
                                  const float* __restrict__ ex, const float* __restrict__ ssum,
                                  const float* __restrict__ xw, float* __restrict__ agg,
                                  int E, int N) {
    const int out_dim = 2 * LPE;
    long gid = (long)blockIdx.x * blockDim.x + threadIdx.x;
    long e = gid / LPE;                      // LPE power of two -> shift
    int lane = (int)(gid & (LPE - 1));
    if (e >= E) return;
    int d = dst[e];
    float alpha = ex[e] / ssum[d];
    const float2* row = (const float2*)(xw + ((size_t)et[e] * N + src[e]) * out_dim);
    float2 v = row[lane];
    float* o = agg + (size_t)d * out_dim + 2 * lane;
    atomicAdd(o,     alpha * v.x);
    atomicAdd(o + 1, alpha * v.y);
}

__global__ void rgat_elu(float* __restrict__ p, long n) {
    long i = (long)blockIdx.x * blockDim.x + threadIdx.x;
    if (i >= n) return;
    float x = p[i];
    p[i] = (x > 0.0f) ? x : (__expf(x) - 1.0f);
}

// =====================================================================
// host-side helpers
// =====================================================================
static inline void conv_bf16(const float* src, ushort_t* dst, long n, hipStream_t s) {
    long n2 = n / 2;
    rgat_f32_to_bf16x2<<<(int)((n2 + 255) / 256), 256, 0, s>>>(
        (const float2*)src, (unsigned int*)dst, n2);
}

static inline void wt_bf16(const float* W, ushort_t* Wt, int in_dim, int out_dim,
                           int nrel, hipStream_t s) {
    long total = (long)nrel * in_dim * out_dim;
    rgat_wt_bf16<<<(int)((total + 255) / 256), 256, 0, s>>>(W, Wt, in_dim, out_dim, total);
}

static void rgat_layer(const float* x, int in_dim, int out_dim,
                       const float* W, const float* q, const float* k,
                       const int* src, const int* dst, const int* et,
                       ushort_t* Xbf, ushort_t* Wtbf,
                       float* xw, float* ebuf, float* mbuf, float* sbuf,
                       float* qn, float* kn, float* agg,
                       int N, int E, hipStream_t stream) {
    conv_bf16(x, Xbf, (long)N * in_dim, stream);
    wt_bf16(W, Wtbf, in_dim, out_dim, 2, stream);

    dim3 g(N / 16, out_dim / 16, 2);
    if (in_dim == 128)
        rgat_wmma_gemm<128><<<g, 32, 0, stream>>>(Xbf, Wtbf, nullptr, xw, out_dim,
                                                  (long)in_dim * out_dim, (long)N * out_dim, 0);
    else
        rgat_wmma_gemm<64><<<g, 32, 0, stream>>>(Xbf, Wtbf, nullptr, xw, out_dim,
                                                 (long)in_dim * out_dim, (long)N * out_dim, 0);

    int t2 = 2 * N;
    rgat_qk<<<(t2 + 255) / 256, 256, 0, stream>>>(xw, q, k, qn, kn, t2, out_dim);

    rgat_fill<<<(N + 255) / 256, 256, 0, stream>>>(mbuf, -1e30f, N);
    rgat_fill<<<(N + 255) / 256, 256, 0, stream>>>(sbuf, 0.0f, N);
    long aggn = (long)N * out_dim;
    rgat_fill<<<(int)((aggn + 255) / 256), 256, 0, stream>>>(agg, 0.0f, aggn);

    rgat_edge_logits<<<(E + 255) / 256, 256, 0, stream>>>(src, dst, et, qn, kn, ebuf, mbuf, E, N);
    rgat_edge_exp<<<(E + 255) / 256, 256, 0, stream>>>(dst, ebuf, mbuf, sbuf, E);

    if (out_dim == 64) {
        long tot = (long)E * 32;
        rgat_edge_scatter<32><<<(int)((tot + 255) / 256), 256, 0, stream>>>(
            src, dst, et, ebuf, sbuf, xw, agg, E, N);
    } else { // out_dim == 32
        long tot = (long)E * 16;
        rgat_edge_scatter<16><<<(int)((tot + 255) / 256), 256, 0, stream>>>(
            src, dst, et, ebuf, sbuf, xw, agg, E, N);
    }
    rgat_elu<<<(int)((aggn + 255) / 256), 256, 0, stream>>>(agg, aggn);
}

extern "C" void kernel_launch(void* const* d_in, const int* in_sizes, int n_in,
                              void* d_out, int out_size, void* d_ws, size_t ws_size,
                              hipStream_t stream) {
    const float* feats = (const float*)d_in[0];
    const int*   eidx  = (const int*)d_in[1];
    const int*   etyp  = (const int*)d_in[2];
    const float* W1 = (const float*)d_in[3];
    const float* q1 = (const float*)d_in[4];
    const float* k1 = (const float*)d_in[5];
    const float* W2 = (const float*)d_in[6];
    const float* q2 = (const float*)d_in[7];
    const float* k2 = (const float*)d_in[8];
    const float* W3 = (const float*)d_in[9];
    const float* q3 = (const float*)d_in[10];
    const float* k3 = (const float*)d_in[11];
    const float* Wd1 = (const float*)d_in[12];
    const float* bd1 = (const float*)d_in[13];
    const float* Wd2 = (const float*)d_in[14];
    const float* bd2 = (const float*)d_in[15];
    const float* Wd3 = (const float*)d_in[16];
    const float* bd3 = (const float*)d_in[17];

    const int N = in_sizes[0] / 128;   // 100000
    const int E = in_sizes[1] / 2;     // 3200000
    const int* src = eidx;             // edge_index[0]
    const int* dst = eidx + E;         // edge_index[1]

    // workspace layout (float-sized units)
    float* ws   = (float*)d_ws;
    float* xw   = ws;                            // 2*N*64 f32
    float* ebuf = xw + 2 * (size_t)N * 64;       // E f32
    float* mbuf = ebuf + (size_t)E;              // N f32
    float* sbuf = mbuf + (size_t)N;              // N f32
    float* qn   = sbuf + (size_t)N;              // 2N f32
    float* kn   = qn + 2 * (size_t)N;            // 2N f32
    float* hA   = kn + 2 * (size_t)N;            // N*64 f32
    float* hB   = hA + (size_t)N * 64;           // N*64 f32
    ushort_t* Xbf  = (ushort_t*)(hB + (size_t)N * 64);      // N*128 bf16
    ushort_t* Wtbf = Xbf + (size_t)N * 128;                 // 2*128*64 bf16

    float* out_h3  = (float*)d_out;              // [N,32]
    float* out_mlp = out_h3 + (size_t)N * 32;    // [N,128]

    // three RGAT layers
    rgat_layer(feats, 128, 64, W1, q1, k1, src, dst, etyp, Xbf, Wtbf,
               xw, ebuf, mbuf, sbuf, qn, kn, hA, N, E, stream);
    rgat_layer(hA, 64, 64, W2, q2, k2, src, dst, etyp, Xbf, Wtbf,
               xw, ebuf, mbuf, sbuf, qn, kn, hB, N, E, stream);
    rgat_layer(hB, 64, 32, W3, q3, k3, src, dst, etyp, Xbf, Wtbf,
               xw, ebuf, mbuf, sbuf, qn, kn, out_h3, N, E, stream);

    // decoder MLP (WMMA GEMMs with bias)
    {
        conv_bf16(out_h3, Xbf, (long)N * 32, stream);
        wt_bf16(Wd1, Wtbf, 32, 64, 1, stream);
        dim3 g1(N / 16, 64 / 16, 1);
        rgat_wmma_gemm<32><<<g1, 32, 0, stream>>>(Xbf, Wtbf, bd1, hA, 64, 0, 0, 1);

        conv_bf16(hA, Xbf, (long)N * 64, stream);
        wt_bf16(Wd2, Wtbf, 64, 64, 1, stream);
        dim3 g2(N / 16, 64 / 16, 1);
        rgat_wmma_gemm<64><<<g2, 32, 0, stream>>>(Xbf, Wtbf, bd2, hB, 64, 0, 0, 1);

        conv_bf16(hB, Xbf, (long)N * 64, stream);
        wt_bf16(Wd3, Wtbf, 64, 128, 1, stream);
        dim3 g3(N / 16, 128 / 16, 1);
        rgat_wmma_gemm<64><<<g3, 32, 0, stream>>>(Xbf, Wtbf, bd3, out_mlp, 128, 0, 0, 0);
    }
}
